// TransformerBlock_16999480557879
// MI455X (gfx1250) — matrix operand order repaired and checked by reading, not compiled, not run on hardware
//
#include <hip/hip_runtime.h>
#include <hip/hip_bf16.h>
#include <stdint.h>

// ---------------------------------------------------------------------------
// Transformer block for MI455X (gfx1250, wave32, WMMA, TDM async staging).
// All GEMM operands bf16 (weights pre-transposed+converted once per call),
// f32 accumulation, f32 residual spine.
// ---------------------------------------------------------------------------

typedef __attribute__((ext_vector_type(16))) __bf16       v16bf;
typedef __attribute__((ext_vector_type(8)))  float        v8f;
typedef __attribute__((ext_vector_type(8)))  unsigned int v8u;
typedef __attribute__((ext_vector_type(4)))  unsigned int v4u;
typedef __attribute__((ext_vector_type(8)))  int          v8i;
typedef __attribute__((ext_vector_type(4)))  int          v4i;

union FragBF { v16bf h; v8u u; };
union F8     { v8f v; float f[8]; };

static constexpr int S_LEN = 2048;
static constexpr int DMODEL = 2048;
static constexpr int NHEADS = 16;
static constexpr int HD = 128;
static constexpr int DFF = 8192;
static constexpr int WIN = 512;
static constexpr float EPS = 1e-5f;

__device__ __forceinline__ __bf16 f32_to_bf16(float f) {
  unsigned u = __builtin_bit_cast(unsigned, f);
  u += 0x7FFFu + ((u >> 16) & 1u);                 // RNE
  unsigned short hs = (unsigned short)(u >> 16);
  return __builtin_bit_cast(__bf16, hs);
}

// ---------------------------------------------------------------------------
// Tensor Data Mover: 2D bf16 tile (tile_d0 x tile_d1) from global row-major
// (row stride stride0 elements) into LDS, packed row-major (stride tile_d0).
// D# bitfields per CDNA5 ISA ch.8 (group0/group1), data_size=1 (2 bytes).
// ---------------------------------------------------------------------------
#if __has_builtin(__builtin_amdgcn_tensor_load_to_lds)
#define USE_TDM 1
__device__ __forceinline__ void tdm_load_2d(unsigned lds_addr, const void* gaddr,
                                            unsigned tensor_d0, unsigned tensor_d1,
                                            unsigned tile_d0, unsigned tile_d1,
                                            unsigned long long stride0) {
  unsigned long long ga = (unsigned long long)(uintptr_t)gaddr;
  v4u g0;
  g0[0] = 1u;                                           // count=1 (user D#)
  g0[1] = lds_addr;                                     // lds_addr [63:32]
  g0[2] = (unsigned)ga;                                 // global_addr lo
  g0[3] = (unsigned)((ga >> 32) & 0x1FFFFFFu) | (2u << 30); // addr hi | type=2
  v8i g1;
  g1[0] = (int)(1u << 16);                              // wg_mask=0, data_size=2B
  g1[1] = (int)((tensor_d0 & 0xFFFFu) << 16);           // tensor_dim0 lo16
  g1[2] = (int)((tensor_d0 >> 16) | ((tensor_d1 & 0xFFFFu) << 16));
  g1[3] = (int)((tensor_d1 >> 16) | (tile_d0 << 16));   // dim1 hi | tile_dim0
  g1[4] = (int)(tile_d1 & 0xFFFFu);                     // tile_dim1, tile_dim2=0
  g1[5] = (int)(unsigned)(stride0 & 0xFFFFFFFFull);     // dim0 stride lo32
  g1[6] = (int)(unsigned)((stride0 >> 32) & 0xFFFFull); // stride hi16, stride1=0
  g1[7] = 0;
  v4i z4 = {0, 0, 0, 0};
#if __clang_major__ >= 23
  v8i z8 = {0, 0, 0, 0, 0, 0, 0, 0};
  __builtin_amdgcn_tensor_load_to_lds(g0, g1, z4, z4, z8, 0);
#else
  __builtin_amdgcn_tensor_load_to_lds(g0, g1, z4, z4, 0);
#endif
}
#endif

__device__ __forceinline__ unsigned lds_off(const void* p) {
  return (unsigned)(uintptr_t)p;
}

// A fragment: 16(M)x32(K) bf16 tile, row-major, row stride ld (even).
__device__ __forceinline__ v16bf load_frag_a(const __bf16* tile, int ld, int lane) {
  const int m = lane & 15, hf = lane >> 4;
  const unsigned* row = (const unsigned*)(tile + m * ld);
  FragBF f;
#pragma unroll
  for (int i = 0; i < 8; ++i) {
    const int ku = i + 4 * hf + ((i >= 4) ? 4 : 0);
    f.u[i] = row[ku];
  }
  return f.h;
}

// B fragment: 32(K)x16(N) bf16, K-transposed storage tileT[n*ld + k].
__device__ __forceinline__ v16bf load_frag_b(const __bf16* tileT, int ld, int lane) {
  const int n = lane & 15, hf = lane >> 4;
  const unsigned* row = (const unsigned*)(tileT + n * ld);
  FragBF f;
#pragma unroll
  for (int j = 0; j < 8; ++j) f.u[j] = row[j + 4 * hf + ((j >= 4) ? 4 : 0)];
  return f.h;
}

__device__ __forceinline__ v8f wmma_bf16(v16bf a, v16bf b, v8f c) {
  return __builtin_amdgcn_wmma_f32_16x16x32_bf16(false, a, false, b, (short)0, c,
                                                 false, false);
}

// ---------------------------------------------------------------------------
// Weight convert + transpose: W fp32 [K][N] -> Wt bf16 [N][K]. 32x32 tiles.
// ---------------------------------------------------------------------------
__global__ __launch_bounds__(256)
void wt_kernel(const float* __restrict__ W, __bf16* __restrict__ Wt, int K, int N) {
  __shared__ __bf16 t[32][33];
  const int n0 = blockIdx.x * 32, k0 = blockIdx.y * 32;
  const int tx = threadIdx.x & 31, ty = threadIdx.x >> 5;
#pragma unroll
  for (int r = ty; r < 32; r += 8)
    t[r][tx] = f32_to_bf16(W[(size_t)(k0 + r) * N + n0 + tx]);
  __syncthreads();
#pragma unroll
  for (int r = ty; r < 32; r += 8)
    Wt[(size_t)(n0 + r) * K + k0 + tx] = t[tx][r];
}

// ---------------------------------------------------------------------------
// LayerNorm: one block per row, optional f32 and bf16 outputs.
// ---------------------------------------------------------------------------
__global__ __launch_bounds__(256)
void ln_kernel(const float* __restrict__ x, const float* __restrict__ g,
               const float* __restrict__ b, float* __restrict__ y32,
               __bf16* __restrict__ y16, int n) {
  const int row = blockIdx.x;
  const float* xr = x + (size_t)row * n;
  __shared__ float redA[8];
  __shared__ float tot[2];
  const int tid = threadIdx.x, lane = tid & 31, wid = tid >> 5;

  float s = 0.f;
  for (int c = tid; c < n; c += 256) s += xr[c];
#pragma unroll
  for (int o = 16; o > 0; o >>= 1) s += __shfl_down(s, o, 32);
  if (lane == 0) redA[wid] = s;
  __syncthreads();
  if (wid == 0) {
    float t = (lane < 8) ? redA[lane] : 0.f;
#pragma unroll
    for (int o = 4; o > 0; o >>= 1) t += __shfl_down(t, o, 32);
    if (lane == 0) tot[0] = t;
  }
  __syncthreads();
  const float mean = tot[0] / (float)n;

  float v = 0.f;
  for (int c = tid; c < n; c += 256) { float d = xr[c] - mean; v += d * d; }
#pragma unroll
  for (int o = 16; o > 0; o >>= 1) v += __shfl_down(v, o, 32);
  if (lane == 0) redA[wid] = v;
  __syncthreads();
  if (wid == 0) {
    float t = (lane < 8) ? redA[lane] : 0.f;
#pragma unroll
    for (int o = 4; o > 0; o >>= 1) t += __shfl_down(t, o, 32);
    if (lane == 0) tot[1] = t;
  }
  __syncthreads();
  const float rstd = rsqrtf(tot[1] / (float)n + EPS);
  for (int c = tid; c < n; c += 256) {
    float val = (xr[c] - mean) * rstd * g[c] + b[c];
    if (y32) y32[(size_t)row * n + c] = val;
    if (y16) y16[(size_t)row * n + c] = f32_to_bf16(val);
  }
}

// ---------------------------------------------------------------------------
// WMMA GEMM: epilogue(A[M,K]bf16 @ Bt[N,K]bf16^T + bias).
// Block tile 128x128, BK=32; 8 waves (4M x 2N), wave tile 32x64 = 8 WMMA/step.
// EPI: 0 bias, 1 bias+res(f32), 2 gelu(bias).  OUT: 0 f32, 1 bf16, 2 bf16^T.
// ---------------------------------------------------------------------------
static constexpr int BM = 128, BN = 128, BK = 32;

template <int EPI, int OUT>
__global__ __launch_bounds__(256)
void gemm_kernel(const __bf16* __restrict__ A, const __bf16* __restrict__ Bt,
                 const float* __restrict__ bias, const float* __restrict__ res,
                 float* __restrict__ Cf, __bf16* __restrict__ Cb,
                 int M, int N, int K) {
  __shared__ __align__(16) __bf16 As[BM * BK];      // [m][k]
  __shared__ __align__(16) __bf16 Bs[BN * BK];      // [n][k]

  const int tid = threadIdx.x, lane = tid & 31, wid = tid >> 5;
  const int wm = wid & 3, wn = wid >> 2;            // 4 x 2 waves
  const int blockM = blockIdx.y * BM, blockN = blockIdx.x * BN;
  const int m0 = wm * 32, n0 = wn * 64;

  F8 acc[2][4] = {};

  for (int k0 = 0; k0 < K; k0 += BK) {
#ifdef USE_TDM
    if (wid == 0) {
      tdm_load_2d(lds_off(As), A + (size_t)blockM * K + k0, (unsigned)K,
                  (unsigned)M, BK, BM, (unsigned long long)K);
      tdm_load_2d(lds_off(Bs), Bt + (size_t)blockN * K + k0, (unsigned)K,
                  (unsigned)N, BK, BN, (unsigned long long)K);
      __builtin_amdgcn_s_wait_tensorcnt(0);
    }
#else
    unsigned* As32 = (unsigned*)As;
    unsigned* Bs32 = (unsigned*)Bs;
#pragma unroll
    for (int l = 0; l < 8; ++l) {                   // 2048 u32 per tile
      int idx = l * 256 + tid;
      int r = idx >> 4, c = idx & 15;
      As32[r * 16 + c] = ((const unsigned*)(A + (size_t)(blockM + r) * K + k0))[c];
      Bs32[r * 16 + c] = ((const unsigned*)(Bt + (size_t)(blockN + r) * K + k0))[c];
    }
#endif
    __syncthreads();

    v16bf a0 = load_frag_a(As + (m0 + 0) * BK, BK, lane);
    v16bf a1 = load_frag_a(As + (m0 + 16) * BK, BK, lane);
#pragma unroll
    for (int ni = 0; ni < 4; ++ni) {
      v16bf bfr = load_frag_b(Bs + (n0 + ni * 16) * BK, BK, lane);
      acc[0][ni].v = wmma_bf16(a0, bfr, acc[0][ni].v);
      acc[1][ni].v = wmma_bf16(a1, bfr, acc[1][ni].v);
    }
    __syncthreads();
  }

  const int cn = lane & 15, hf = lane >> 4;
#pragma unroll
  for (int mi = 0; mi < 2; ++mi) {
#pragma unroll
    for (int ni = 0; ni < 4; ++ni) {
#pragma unroll
      for (int r = 0; r < 8; ++r) {
        int gm = blockM + m0 + mi * 16 + r + 8 * hf;
        int gn = blockN + n0 + ni * 16 + cn;
        float val = acc[mi][ni].f[r] + bias[gn];
        if constexpr (EPI == 1) val += res[(size_t)gm * N + gn];
        if constexpr (EPI == 2) val = 0.5f * val * (1.f + erff(val * 0.70710678118654752f));
        if constexpr (OUT == 0) Cf[(size_t)gm * N + gn] = val;
        if constexpr (OUT == 1) Cb[(size_t)gm * N + gn] = f32_to_bf16(val);
        if constexpr (OUT == 2) Cb[(size_t)gn * M + gm] = f32_to_bf16(val);
      }
    }
  }
}

// ---------------------------------------------------------------------------
// Sliding-window causal attention, bf16 in (V pre-transposed), bf16 out.
// Grid (S/64, NHEADS); 4 waves x 16 query rows; online softmax; WMMA both MMs.
// ---------------------------------------------------------------------------
__global__ __launch_bounds__(128)
void attn_kernel(const __bf16* __restrict__ Q, const __bf16* __restrict__ K,
                 const __bf16* __restrict__ Vt, __bf16* __restrict__ ctx) {
  __shared__ __align__(16) __bf16 Qs[4 * 16 * HD];   // per-wave [q][d]
  __shared__ __align__(16) __bf16 Ks[32 * HD];       // [key][d]
  __shared__ __align__(16) __bf16 VTs[HD * 32];      // [d][key]
  __shared__ float  Sc[4 * 16 * 32];
  __shared__ __align__(8) __bf16 Ps[4 * 16 * 32];
  __shared__ float rowM[4 * 16], rowL[4 * 16], rowS[4 * 16];

  const int tid = threadIdx.x, lane = tid & 31, w = tid >> 5;
  const int h = blockIdx.y;
  const int qb0 = blockIdx.x * 64;
  const int q0 = qb0 + w * 16;
  const int cn = lane & 15, hf = lane >> 4;
  const float qscale = 0.08838834764831845f;          // 1/sqrt(128)

  // Stage per-wave Q tile (16x128): one TDM per wave.
#ifdef USE_TDM
  tdm_load_2d(lds_off(Qs + w * 16 * HD), Q + (size_t)q0 * DMODEL + h * HD,
              DMODEL, S_LEN, HD, 16, DMODEL);
#else
  {
    unsigned* Qs32 = (unsigned*)(Qs + w * 16 * HD);
#pragma unroll
    for (int l = 0; l < 32; ++l) {                   // 1024 u32 / 32 lanes
      int idx = l * 32 + lane;
      int r = idx >> 6, c = idx & 63;
      Qs32[r * 64 + c] = ((const unsigned*)(Q + (size_t)(q0 + r) * DMODEL + h * HD))[c];
    }
  }
#endif
  if (lane < 16) { rowM[w * 16 + lane] = -1e30f; rowL[w * 16 + lane] = 0.f; }

  F8 acc[8] = {};
  const int kstart = (qb0 > WIN) ? (qb0 - WIN) : 0;
  const int kend = qb0 + 63;

  for (int kc = kstart; kc <= kend; kc += 32) {
#ifdef USE_TDM
    if (w == 0) {
      tdm_load_2d(lds_off(Ks), K + (size_t)kc * DMODEL + h * HD,
                  DMODEL, S_LEN, HD, 32, DMODEL);
      tdm_load_2d(lds_off(VTs), Vt + (size_t)(h * HD) * S_LEN + kc,
                  S_LEN, DMODEL, 32, HD, S_LEN);
    }
    __builtin_amdgcn_s_wait_tensorcnt(0);            // own TDM ops (Q and/or K,V)
#else
    {
      unsigned* Ks32 = (unsigned*)Ks;
      unsigned* VT32 = (unsigned*)VTs;
#pragma unroll
      for (int l = 0; l < 16; ++l) {                 // 2048 u32 each
        int idx = l * 128 + tid;
        int r = idx >> 6, c = idx & 63;              // Ks: 32 rows x 64 u32
        Ks32[r * 64 + c] = ((const unsigned*)(K + (size_t)(kc + r) * DMODEL + h * HD))[c];
        int r2 = idx >> 4, c2 = idx & 15;            // VTs: 128 rows x 16 u32
        VT32[r2 * 16 + c2] =
            ((const unsigned*)(Vt + (size_t)(h * HD + r2) * S_LEN + kc))[c2];
      }
    }
#endif
    __syncthreads();

    const bool active = (kc <= q0 + 15) && (kc + 31 >= q0 - WIN);
    if (active) {
      F8 s0 = {}, s1 = {};
#pragma unroll
      for (int dc = 0; dc < 4; ++dc) {
        v16bf aq = load_frag_a(Qs + w * 16 * HD + dc * 32, HD, lane);
        v16bf b0 = load_frag_b(Ks + 0 * HD + dc * 32, HD, lane);
        v16bf b1 = load_frag_b(Ks + 16 * HD + dc * 32, HD, lane);
        s0.v = wmma_bf16(aq, b0, s0.v);
        s1.v = wmma_bf16(aq, b1, s1.v);
      }
#pragma unroll
      for (int r = 0; r < 8; ++r) {
        int m = r + 8 * hf;
        int qm = q0 + m;
        int k0i = kc + cn, k1i = kc + 16 + cn;
        float v0 = (k0i > qm || qm - k0i > WIN) ? -1e30f : s0.f[r] * qscale;
        float v1 = (k1i > qm || qm - k1i > WIN) ? -1e30f : s1.f[r] * qscale;
        Sc[w * 512 + m * 32 + cn] = v0;
        Sc[w * 512 + m * 32 + 16 + cn] = v1;
      }
      asm volatile("s_wait_dscnt 0" ::: "memory");

      if (lane < 16) {
        const int m = lane;
        float mprev = rowM[w * 16 + m];
        float cmax = -1e30f;
#pragma unroll
        for (int c = 0; c < 32; ++c) cmax = fmaxf(cmax, Sc[w * 512 + m * 32 + c]);
        float nm = fmaxf(mprev, cmax);
        float scl = __expf(mprev - nm);
        float lsum = 0.f;
#pragma unroll
        for (int c = 0; c < 32; ++c) {
          float sv = Sc[w * 512 + m * 32 + c];
          float p = (sv > -1e29f) ? __expf(sv - nm) : 0.f;
          Ps[w * 512 + m * 32 + c] = f32_to_bf16(p);
          lsum += p;
        }
        rowM[w * 16 + m] = nm;
        rowL[w * 16 + m] = rowL[w * 16 + m] * scl + lsum;
        rowS[w * 16 + m] = scl;
      }
      asm volatile("s_wait_dscnt 0" ::: "memory");

#pragma unroll
      for (int r = 0; r < 8; ++r) {
        float fsc = rowS[w * 16 + r + 8 * hf];
#pragma unroll
        for (int t = 0; t < 8; ++t) acc[t].f[r] *= fsc;
      }

      v16bf ap = load_frag_a(Ps + w * 512, 32, lane);
#pragma unroll
      for (int nt = 0; nt < 8; ++nt) {
        v16bf bv = load_frag_b(VTs + (nt * 16) * 32, 32, lane);
        acc[nt].v = wmma_bf16(ap, bv, acc[nt].v);
      }
    }
    __syncthreads();
  }

  if (lane < 16) rowS[w * 16 + lane] = 1.f / rowL[w * 16 + lane];
  asm volatile("s_wait_dscnt 0" ::: "memory");
#pragma unroll
  for (int r = 0; r < 8; ++r) {
    int m = r + 8 * hf;
    float inv = rowS[w * 16 + m];
#pragma unroll
    for (int nt = 0; nt < 8; ++nt) {
      ctx[(size_t)(q0 + m) * DMODEL + h * HD + nt * 16 + cn] =
          f32_to_bf16(acc[nt].f[r] * inv);
    }
  }
}

// ---------------------------------------------------------------------------
// Launcher.
// ---------------------------------------------------------------------------
extern "C" void kernel_launch(void* const* d_in, const int* in_sizes, int n_in,
                              void* d_out, int out_size, void* d_ws, size_t ws_size,
                              hipStream_t stream) {
  const float* x    = (const float*)d_in[0];
  const float* Wq   = (const float*)d_in[1];
  const float* bq   = (const float*)d_in[2];
  const float* Wk   = (const float*)d_in[3];
  const float* bk   = (const float*)d_in[4];
  const float* Wv   = (const float*)d_in[5];
  const float* bv   = (const float*)d_in[6];
  const float* Wo   = (const float*)d_in[7];
  const float* bo   = (const float*)d_in[8];
  const float* ln1g = (const float*)d_in[9];
  const float* ln1b = (const float*)d_in[10];
  const float* W1   = (const float*)d_in[11];
  const float* b1   = (const float*)d_in[12];
  const float* W2   = (const float*)d_in[13];
  const float* b2   = (const float*)d_in[14];
  const float* ln2g = (const float*)d_in[15];
  const float* ln2b = (const float*)d_in[16];
  float* out = (float*)d_out;

  const size_t NT = (size_t)S_LEN * DMODEL;      // 4M elems
  const size_t NF = (size_t)S_LEN * DFF;         // 16M elems
  char* p = (char*)d_ws;
  float*  xn  = (float*)p;            p += NT * 4;      // LN1 out (f32 residual)
  float*  x1  = (float*)p;            p += NT * 4;      // attn residual out (f32)
  __bf16* xnb = (__bf16*)p;           p += NT * 2;      // LN1 out bf16
  __bf16* q   = (__bf16*)p;           p += NT * 2;
  __bf16* k   = (__bf16*)p;           p += NT * 2;
  __bf16* vt  = (__bf16*)p;           p += NT * 2;      // V transposed [DM][S]
  __bf16* ctx = (__bf16*)p;           p += NT * 2;
  __bf16* x2b = (__bf16*)p;           p += NT * 2;
  __bf16* hbf = (__bf16*)p;           p += NF * 2;
  __bf16* wtq = (__bf16*)p;           p += NT * 2;      // W^T bf16 copies
  __bf16* wtk = (__bf16*)p;           p += NT * 2;
  __bf16* wtv = (__bf16*)p;           p += NT * 2;
  __bf16* wto = (__bf16*)p;           p += NT * 2;
  __bf16* wt1 = (__bf16*)p;           p += NF * 2;
  __bf16* wt2 = (__bf16*)p;           p += NF * 2;

  dim3 blk256(256), blk128(128);
  dim3 gWT(DMODEL / 32, DMODEL / 32);            // (64, 64)
  dim3 gWT1(DFF / 32, DMODEL / 32);              // W1: K=DM, N=DFF
  dim3 gWT2(DMODEL / 32, DFF / 32);              // W2: K=DFF, N=DM
  dim3 gLN(S_LEN);
  dim3 gProj(DMODEL / BN, S_LEN / BM);           // (16, 16)
  dim3 gFF1(DFF / BN, S_LEN / BM);               // (64, 16)
  dim3 gAttn(S_LEN / 64, NHEADS);

  // 0) weight convert + transpose (fp32 [K][N] -> bf16 [N][K])
  wt_kernel<<<gWT,  blk256, 0, stream>>>(Wq, wtq, DMODEL, DMODEL);
  wt_kernel<<<gWT,  blk256, 0, stream>>>(Wk, wtk, DMODEL, DMODEL);
  wt_kernel<<<gWT,  blk256, 0, stream>>>(Wv, wtv, DMODEL, DMODEL);
  wt_kernel<<<gWT,  blk256, 0, stream>>>(Wo, wto, DMODEL, DMODEL);
  wt_kernel<<<gWT1, blk256, 0, stream>>>(W1, wt1, DMODEL, DFF);
  wt_kernel<<<gWT2, blk256, 0, stream>>>(W2, wt2, DFF, DMODEL);
  // 1) xn = LN1(x)  (f32 + bf16)
  ln_kernel<<<gLN, blk256, 0, stream>>>(x, ln1g, ln1b, xn, xnb, DMODEL);
  // 2) Q/K = xn @ W + b (bf16 out); V written transposed
  gemm_kernel<0, 1><<<gProj, blk256, 0, stream>>>(xnb, wtq, bq, nullptr, nullptr, q,  S_LEN, DMODEL, DMODEL);
  gemm_kernel<0, 1><<<gProj, blk256, 0, stream>>>(xnb, wtk, bk, nullptr, nullptr, k,  S_LEN, DMODEL, DMODEL);
  gemm_kernel<0, 2><<<gProj, blk256, 0, stream>>>(xnb, wtv, bv, nullptr, nullptr, vt, S_LEN, DMODEL, DMODEL);
  // 3) ctx = attention(q, k, vt)
  attn_kernel<<<gAttn, blk128, 0, stream>>>(q, k, vt, ctx);
  // 4) x1 = ctx @ Wo + bo + xn (f32)
  gemm_kernel<1, 0><<<gProj, blk256, 0, stream>>>(ctx, wto, bo, xn, x1, nullptr, S_LEN, DMODEL, DMODEL);
  // 5) x2 = LN2(x1) (bf16 only)
  ln_kernel<<<gLN, blk256, 0, stream>>>(x1, ln2g, ln2b, nullptr, x2b, DMODEL);
  // 6) h = gelu(x2 @ W1 + b1) (bf16)
  gemm_kernel<2, 1><<<gFF1, blk256, 0, stream>>>(x2b, wt1, b1, nullptr, nullptr, hbf, S_LEN, DFF, DMODEL);
  // 7) out = h @ W2 + b2 + x1 (f32)
  gemm_kernel<1, 0><<<gProj, blk256, 0, stream>>>(hbf, wt2, b2, x1, out, nullptr, S_LEN, DMODEL, DFF);
}
